// DotProductAttention_11940009083379
// MI455X (gfx1250) — matrix operand order repaired and checked by reading, not compiled, not run on hardware
//
#include <hip/hip_runtime.h>
#include <hip/hip_bf16.h>

typedef __attribute__((ext_vector_type(16))) _Float16 v16h;
typedef __attribute__((ext_vector_type(8)))  float    v8f;

#define WAVES 8
#define QT    16              // query rows per wave
#define QBLK  (WAVES * QT)    // 128 query rows per block
#define KT    32              // keys per flash iteration (= WMMA K for the PV GEMM)
#define DHEAD 128

// Flash attention, fp32 in/out, f16 WMMA compute with fp32 accumulation.
// Computes S^T = K_tile x Q^T so softmax stats are per-lane and the C-layout
// of S^T feeds the P x V WMMA's A operand with no cross-lane movement.
// Tiles fully past valid_len are skipped: their exp() terms underflow to
// exactly 0.0f in fp32, identical to the reference (vl==0 runs the full loop).
__global__ __launch_bounds__(WAVES * 32)
void DotProductAttention_fa_kernel(const float* __restrict__ Qg,
                                   const float* __restrict__ Kg,
                                   const float* __restrict__ Vg,
                                   const int*   __restrict__ VL,
                                   float* __restrict__ Og,
                                   int B, int Qlen, int Klen)
{
    __shared__ _Float16 s_k [KT * DHEAD];   // [key][d]   row-major, 8 KB
    __shared__ _Float16 s_vt[DHEAD * KT];   // [d][key]   transposed, 8 KB

    const int tid  = threadIdx.x;
    const int w    = tid >> 5;
    const int lane = tid & 31;
    const int ln   = lane & 15;     // n index inside a 16-wide tile (here: query)
    const int half = lane >> 4;     // lane half selector

    const int b  = blockIdx.y;
    const int q0 = blockIdx.x * QBLK;
    const int vl = __builtin_amdgcn_readfirstlane(VL[b]);   // wave-uniform SGPR

    const float scale = 0.08838834764831845f;   // 1/sqrt(128)

    // ---- Q as B-operand (loaded once): b[j] <-> k = 32*dt + 16*half + j, n = q = ln
    v16h qb[4];
    {
        const float* qrow = Qg + ((size_t)b * Qlen + q0 + w * QT + ln) * DHEAD;
        #pragma unroll
        for (int dt = 0; dt < 4; ++dt) {
            const int d0 = 32 * dt + 16 * half;
            #pragma unroll
            for (int j = 0; j < 16; ++j)
                qb[dt][j] = (_Float16)qrow[d0 + j];
        }
    }

    // ---- O accumulator: 8 n-tiles over d=128, C-layout (lane = d%16, VGPR r = q row r+8*half)
    v8f o[8];
    #pragma unroll
    for (int nt = 0; nt < 8; ++nt)
        #pragma unroll
        for (int r = 0; r < 8; ++r) o[nt][r] = 0.0f;

    float m_run = -1e30f;   // running row max (per lane == per query)
    float l_run = 0.0f;     // running denominator

    const float* kp0 = Kg + (size_t)b * Klen * DHEAD;
    const float* vp0 = Vg + (size_t)b * Klen * DHEAD;
    const int niter_full = Klen / KT;
    // Keys >= vl give exp terms that underflow to exactly 0.0f (fp32), same as
    // the reference, so only ceil(vl/KT) tiles matter -- except vl==0, where
    // the reference softmax is uniform over ALL keys and we must run them all.
    const int niter = (vl > 0) ? min(niter_full, (vl + KT - 1) / KT) : niter_full;

    for (int it = 0; it < niter; ++it) {
        const int kbase = it * KT;
        const float* kp = kp0 + (size_t)kbase * DHEAD;
        const float* vp = vp0 + (size_t)kbase * DHEAD;

        __syncthreads();   // previous iteration's readers done before refill
        // Cooperative stage: K row-major, V transposed; fp32 -> f16
        for (int i = tid; i < KT * DHEAD; i += WAVES * 32) {
            const int key = i >> 7;     // i / 128
            const int d   = i & 127;
            s_k[i]              = (_Float16)kp[i];
            s_vt[d * KT + key]  = (_Float16)vp[i];
        }
        if (it + 1 < niter) {   // speculative prefetch of next tiles (global_prefetch_b8)
            __builtin_prefetch(kp + KT * DHEAD + tid * 16, 0, 0);
            __builtin_prefetch(vp + KT * DHEAD + tid * 16, 0, 0);
        }
        __syncthreads();

        // ---- S^T = K_tile x Q^T : two 16-key subtiles, K-dim = d (4 x 32)
        v8f c0, c1;
        #pragma unroll
        for (int r = 0; r < 8; ++r) { c0[r] = 0.0f; c1[r] = 0.0f; }
        #pragma unroll
        for (int dt = 0; dt < 4; ++dt) {
            // A-layout: lane m = key (= ln), a[j] <-> d = 32*dt + 16*half + j
            const int d0 = 32 * dt + 16 * half;
            const _Float16* k0 = &s_k[(ln)      * DHEAD];
            const _Float16* k1 = &s_k[(16 + ln) * DHEAD];
            v16h a0, a1;
            #pragma unroll
            for (int j = 0; j < 8; ++j) {
                a0[j]     = k0[d0 + j];
                a0[j + 8] = k0[d0 + 8 + j];
                a1[j]     = k1[d0 + j];
                a1[j + 8] = k1[d0 + 8 + j];
            }
            c0 = __builtin_amdgcn_wmma_f32_16x16x32_f16(false, a0, false, qb[dt],
                                                        (short)0, c0, false, false);
            c1 = __builtin_amdgcn_wmma_f32_16x16x32_f16(false, a1, false, qb[dt],
                                                        (short)0, c1, false, false);
        }

        // ---- scale (+ mask only in the boundary tile; condition is scalar)
        float p0[8], p1[8];
        #pragma unroll
        for (int r = 0; r < 8; ++r) { p0[r] = c0[r] * scale; p1[r] = c1[r] * scale; }
        if (kbase + KT > vl) {          // wave-uniform branch
            #pragma unroll
            for (int r = 0; r < 8; ++r) {
                const int ki = kbase + r + 8 * half;    // key of c0[r]; c1: +16
                if (ki      >= vl) p0[r] = -1e6f;
                if (ki + 16 >= vl) p1[r] = -1e6f;
            }
        }

        float tm = -1e30f;
        #pragma unroll
        for (int r = 0; r < 8; ++r) tm = fmaxf(tm, fmaxf(p0[r], p1[r]));
        tm = fmaxf(tm, __shfl_xor(tm, 16, 32));          // combine lane halves

        const float m_new = fmaxf(m_run, tm);
        const float alpha = __expf(m_run - m_new);
        float rs = 0.0f;
        #pragma unroll
        for (int r = 0; r < 8; ++r) {
            p0[r] = __expf(p0[r] - m_new);
            p1[r] = __expf(p1[r] - m_new);
            rs += p0[r] + p1[r];
        }
        rs += __shfl_xor(rs, 16, 32);
        l_run = l_run * alpha + rs;
        m_run = m_new;

        // ---- rescale O: row of o[*][r] is q = r + 8*half -> gather that row's alpha
        #pragma unroll
        for (int r = 0; r < 8; ++r) {
            const float ar = __shfl(alpha, r + 8 * half, 32);
            #pragma unroll
            for (int nt = 0; nt < 8; ++nt) o[nt][r] *= ar;
        }

        // ---- P (A-layout, free): a[j] = p0[j], a[j+8] = p1[j]
        v16h pa;
        #pragma unroll
        for (int j = 0; j < 8; ++j) {
            pa[j]     = (_Float16)p0[j];
            pa[j + 8] = (_Float16)p1[j];
        }

        // ---- O += P x V : B-operand from transposed V, b[j] <-> key = 8*half + (j<8 ? j : j+8)
        #pragma unroll
        for (int nt = 0; nt < 8; ++nt) {
            const _Float16* vtp = &s_vt[(16 * nt + ln) * KT + 8 * half];
            v16h vb;
            #pragma unroll
            for (int j = 0; j < 16; ++j) vb[j] = vtp[j < 8 ? j : j + 8];
            o[nt] = __builtin_amdgcn_wmma_f32_16x16x32_f16(false, pa, false, vb,
                                                           (short)0, o[nt], false, false);
        }
    }

    // ---- epilogue: normalize by row sum and store
    const float linv = 1.0f / l_run;
    float* obase = Og + ((size_t)b * Qlen + q0 + w * QT) * DHEAD;
    #pragma unroll
    for (int r = 0; r < 8; ++r) {
        const float lr = __shfl(linv, r + 8 * half, 32);
        float* orow = obase + (size_t)(r + 8 * half) * DHEAD + ln;
        #pragma unroll
        for (int nt = 0; nt < 8; ++nt)
            orow[16 * nt] = o[nt][r] * lr;
    }
}

extern "C" void kernel_launch(void* const* d_in, const int* in_sizes, int n_in,
                              void* d_out, int out_size, void* d_ws, size_t ws_size,
                              hipStream_t stream) {
    (void)n_in; (void)d_ws; (void)ws_size; (void)out_size;
    const float* Qg = (const float*)d_in[0];
    const float* Kg = (const float*)d_in[1];
    const float* Vg = (const float*)d_in[2];
    const int*   VL = (const int*)d_in[3];
    float* Og = (float*)d_out;

    const int B    = in_sizes[3];
    const int Qlen = in_sizes[0] / (B * DHEAD);
    const int Klen = in_sizes[1] / (B * DHEAD);

    dim3 block(WAVES * 32, 1, 1);
    dim3 grid(Qlen / QBLK, B, 1);
    DotProductAttention_fa_kernel<<<grid, block, 0, stream>>>(Qg, Kg, Vg, VL, Og,
                                                              B, Qlen, Klen);
}
